// TCM_3917010174012
// MI455X (gfx1250) — matrix operand order repaired and checked by reading, not compiled, not run on hardware
//
#include <hip/hip_runtime.h>

typedef __attribute__((ext_vector_type(16))) _Float16 v16h;
typedef __attribute__((ext_vector_type(8)))  float    v8f;
typedef __attribute__((ext_vector_type(2)))  float    v2f;

#define HW    784
#define W28   28
#define BTN   128
#define NSEG  8
#define NCLIP 16
#define PATCH 15
#define NDISP 225

// ---------------------------------------------------------------------------
// 1x1 conv as fp32 WMMA GEMM (V_WMMA_F32_16X16X4_F32).
// A: 16 pixels x 4 ch (lane = row, lanes16-31 hold K=2,3) ; B: 4x16 out-ch.
// Used for the precision-critical 128->64 reduce conv (feeds argmax path).
// grid = (numPixels/16, Cout/16), block = 32 (one wave, EXEC all ones).
// ---------------------------------------------------------------------------
__global__ __launch_bounds__(32)
void pconv_f32wmma_k(const float* __restrict__ in, const float* __restrict__ w,
                     const float* __restrict__ scale, const float* __restrict__ bias,
                     float* __restrict__ out, int Cin, int Cout)
{
  int mt = blockIdx.x, nt = blockIdx.y;
  int lane = threadIdx.x & 31;
  int lr = lane & 15, hi = lane >> 4;
  int pix0  = mt * 16;
  int bt    = pix0 / HW;
  int pbase = pix0 % HW;                       // 784 % 16 == 0: tile stays in frame
  const float* inb  = in + (long)bt * Cin * HW + pbase + lr;
  const float* wrow = w  + (long)(nt * 16 + lr) * Cin;
  v8f acc = {};
  for (int k0 = 0; k0 < Cin; k0 += 4) {
    int ka = k0 + hi * 2;                      // lanes16-31 hold K half 2,3
    v2f a, b;
    a[0] = inb[(long)(ka + 0) * HW];
    a[1] = inb[(long)(ka + 1) * HW];
    b[0] = wrow[ka + 0];
    b[1] = wrow[ka + 1];
    acc = __builtin_amdgcn_wmma_f32_16x16x4_f32(false, a, false, b,
                                                (short)0, acc, false, false);
  }
  int   n  = nt * 16 + lr;
  float s  = scale[n], bb = bias[n];
  float* ob = out + (long)bt * Cout * HW + (long)n * HW + pbase;
#pragma unroll
  for (int r = 0; r < 8; ++r) {                // D: m = r + 8*hi, n = lane&15
    float v = s * acc[r] + bb;
    ob[r + hi * 8] = v > 0.f ? v : 0.f;
  }
}

// ---------------------------------------------------------------------------
// 1x1 conv as f16 WMMA GEMM (codegen-confirmed V_WMMA_F32_16X16X32_F16).
// A layout (16-bit A 16x32): k = 2*(e&8) + 8*hi + (e&7).  B: same k order.
// Used for TAM pointwise stages 6->16, 16->32, 32->64, 64->128.
// ---------------------------------------------------------------------------
__global__ __launch_bounds__(32)
void pconv_f16wmma_k(const float* __restrict__ in, const float* __restrict__ w,
                     const float* __restrict__ scale, const float* __restrict__ bias,
                     float* __restrict__ out, int Cin, int Cout)
{
  int mt = blockIdx.x, nt = blockIdx.y;
  int lane = threadIdx.x & 31;
  int lr = lane & 15, hi = lane >> 4;
  int pix0  = mt * 16;
  int bt    = pix0 / HW;
  int pbase = pix0 % HW;
  const float* inb  = in + (long)bt * Cin * HW + pbase + lr;
  const float* wrow = w  + (long)(nt * 16 + lr) * Cin;
  v8f acc = {};
  for (int k0 = 0; k0 < Cin; k0 += 32) {
    v16h a, b;
#pragma unroll
    for (int e = 0; e < 16; ++e) {
      int ka = k0 + ((e & 8) << 1) + hi * 8 + (e & 7);
      a[e] = (_Float16)((ka < Cin) ? inb[(long)ka * HW] : 0.f);
      int kb = k0 + ((e & 8) << 1) + hi * 8 + (e & 7);
      b[e] = (_Float16)((kb < Cin) ? wrow[kb] : 0.f);
    }
    acc = __builtin_amdgcn_wmma_f32_16x16x32_f16(false, a, false, b,
                                                 (short)0, acc, false, false);
  }
  int   n  = nt * 16 + lr;
  float s  = scale[n], bb = bias[n];
  float* ob = out + (long)bt * Cout * HW + (long)n * HW + pbase;
#pragma unroll
  for (int r = 0; r < 8; ++r) {
    float v = s * acc[r] + bb;
    ob[r + hi * 8] = v > 0.f ? v : 0.f;
  }
}

// --------------------------- channel L2 norm (in place) --------------------
__global__ void l2norm_k(float* __restrict__ xr)
{
  int pix = blockIdx.x * blockDim.x + threadIdx.x;
  if (pix >= BTN * HW) return;
  int bt = pix / HW, p = pix % HW;
  float* base = xr + (long)bt * 64 * HW + p;
  float v[64];
  float ss = 1e-6f;
#pragma unroll
  for (int c = 0; c < 64; ++c) { v[c] = base[(long)c * HW]; ss += v[c] * v[c]; }
  float inv = rsqrtf(ss);
#pragma unroll
  for (int c = 0; c < 64; ++c) base[(long)c * HW] = v[c] * inv;
}

// ----------------- fused correlation + flow_soft (per pixel) ---------------
__device__ __forceinline__ void flow_soft_write(float corr, int t,
                                                float* red, int* redi,
                                                float* dst, bool dup)
{
  // argmax of corr (first-max tie-break, like jnp.argmax); corr >= 0
  red[t]  = (t < NDISP) ? corr : -1.0f;
  redi[t] = t;
  __syncthreads();
  for (int s = 128; s > 0; s >>= 1) {
    if (t < s) {
      float a = red[t], b = red[t + s];
      int ia = redi[t], ib = redi[t + s];
      if (b > a || (b == a && ib < ia)) { red[t] = b; redi[t] = ib; }
    }
    __syncthreads();
  }
  float conf = red[0];
  int   idx  = redi[0];
  __syncthreads();

  int py = t / PATCH, px = t % PATCH;
  float iy = (float)(idx / PATCH), ix = (float)(idx % PATCH);
  float logit = -1.0f;
  if (t < NDISP) {
    float dy = (float)py - iy, dx = (float)px - ix;
    float g = expf(-(dy * dy + dx * dx) * 0.02f);  // 1/(2*sigma^2)=1/50 per axis
    logit = g * corr * 100.0f;                     // TEMP = 100
  }
  red[t] = logit;
  __syncthreads();
  for (int s = 128; s > 0; s >>= 1) {
    if (t < s) red[t] = fmaxf(red[t], red[t + s]);
    __syncthreads();
  }
  float m = red[0];
  __syncthreads();

  float e = (t < NDISP) ? expf(logit - m) : 0.f;
  red[t] = e; __syncthreads();
  for (int s = 128; s > 0; s >>= 1) { if (t < s) red[t] += red[t + s]; __syncthreads(); }
  float ssum = red[0]; __syncthreads();

  red[t] = e * (float)(px - 7); __syncthreads();
  for (int s = 128; s > 0; s >>= 1) { if (t < s) red[t] += red[t + s]; __syncthreads(); }
  float sx = red[0]; __syncthreads();

  red[t] = e * (float)(py - 7); __syncthreads();
  for (int s = 128; s > 0; s >>= 1) { if (t < s) red[t] += red[t + s]; __syncthreads(); }
  float sy = red[0]; __syncthreads();

  if (t == 0) {
    float fx = sx / ssum * (1.0f / 7.0f);
    float fy = sy / ssum * (1.0f / 7.0f);
    dst[0] = fx; dst[HW] = fy; dst[2 * HW] = conf;
    if (dup) {  // frame T-1 duplicated to frame T
      dst[6 * HW] = fx; dst[7 * HW] = fy; dst[8 * HW] = conf;
    }
  }
}

__global__ __launch_bounds__(256)
void corrflow_k(const float* __restrict__ xn, float* __restrict__ out6)
{
  __shared__ float f1a[64], f1b[64];
  __shared__ float red[256];
  __shared__ int   redi[256];
  int p  = blockIdx.x;          // pixel 0..783
  int tp = blockIdx.y;          // pair 0..6 (post frame tp+1)
  int n  = blockIdx.z;          // clip 0..15
  int y = p / W28, x = p % W28;
  int t = threadIdx.x;

  const float* post = xn + (long)(n * NSEG + tp + 1) * 64 * HW;
  const float* pre1 = xn + (long)(n * NSEG + tp)     * 64 * HW;
  const float* pre0 = xn + (long)(n * NSEG)          * 64 * HW;
  if (t < 64) { f1a[t] = pre1[(long)t * HW + p]; f1b[t] = pre0[(long)t * HW + p]; }
  __syncthreads();

  float c1 = 0.f, c2 = 0.f;
  if (t < NDISP) {
    int yy = y + t / PATCH - 7, xx = x + t % PATCH - 7;
    if (yy >= 0 && yy < W28 && xx >= 0 && xx < W28) {
      const float* f2 = post + yy * W28 + xx;
#pragma unroll 8
      for (int c = 0; c < 64; ++c) {
        float v = f2[(long)c * HW];
        c1 += f1a[c] * v;
        c2 += f1b[c] * v;
      }
    }
    c1 = fmaxf(c1, 0.f);     // relu(correlation)
    c2 = fmaxf(c2, 0.f);
  }
  float* base = out6 + ((long)(n * NSEG + tp) * 6) * HW + p;
  flow_soft_write(c1, t, red, redi, base,          tp == 6);  // branch1 -> ch 0..2
  flow_soft_write(c2, t, red, redi, base + 3 * HW, tp == 6);  // branch2 -> ch 3..5
}

// ------------------- grouped 3x3 conv (6ch, groups=3) + BN/ReLU ------------
__global__ void gconv6_k(const float* __restrict__ in, const float* __restrict__ w,
                         const float* __restrict__ s, const float* __restrict__ b,
                         float* __restrict__ out)
{
  long i = blockIdx.x * (long)blockDim.x + threadIdx.x;
  if (i >= (long)BTN * 6 * HW) return;
  int p = (int)(i % HW); long r = i / HW; int oc = (int)(r % 6); int bt = (int)(r / 6);
  int g = oc >> 1;
  int y = p / W28, x = p % W28;
  float acc = 0.f;
#pragma unroll
  for (int j = 0; j < 2; ++j) {
    const float* ip = in + ((long)bt * 6 + (g * 2 + j)) * HW;
    const float* wp = w + (oc * 2 + j) * 9;
#pragma unroll
    for (int ky = 0; ky < 3; ++ky) {
      int yy = y + ky - 1; if (yy < 0 || yy >= W28) continue;
#pragma unroll
      for (int kx = 0; kx < 3; ++kx) {
        int xx = x + kx - 1; if (xx < 0 || xx >= W28) continue;
        acc += wp[ky * 3 + kx] * ip[yy * W28 + xx];
      }
    }
  }
  float v = s[oc] * acc + b[oc];
  out[i] = v > 0.f ? v : 0.f;
}

// ------------------------- depthwise 3x3 conv + BN/ReLU --------------------
__global__ void dwconv_k(const float* __restrict__ in, const float* __restrict__ w,
                         const float* __restrict__ s, const float* __restrict__ b,
                         float* __restrict__ out, int C)
{
  long i = blockIdx.x * (long)blockDim.x + threadIdx.x;
  if (i >= (long)BTN * C * HW) return;
  int p = (int)(i % HW); long r = i / HW; int c = (int)(r % C); int bt = (int)(r / C);
  int y = p / W28, x = p % W28;
  const float* ip = in + ((long)bt * C + c) * HW;
  const float* wp = w + c * 9;
  float acc = 0.f;
#pragma unroll
  for (int ky = 0; ky < 3; ++ky) {
    int yy = y + ky - 1; if (yy < 0 || yy >= W28) continue;
#pragma unroll
    for (int kx = 0; kx < 3; ++kx) {
      int xx = x + kx - 1; if (xx < 0 || xx >= W28) continue;
      acc += wp[ky * 3 + kx] * ip[yy * W28 + xx];
    }
  }
  float v = s[c] * acc + b[c];
  out[i] = v > 0.f ? v : 0.f;
}

// -------------------- ECA: spatial mean per (clip, chan, frame) ------------
__global__ __launch_bounds__(256)
void eca_mean_k(const float* __restrict__ t4, float* __restrict__ ybuf)
{
  __shared__ float red[256];
  int item = blockIdx.x;                    // ((n*128)+c)*8 + t
  int t = item & 7, c = (item >> 3) & 127, n = item >> 10;
  const float* p = t4 + ((long)(n * NSEG + t) * 128 + c) * HW;
  float s = 0.f;
  for (int i = threadIdx.x; i < HW; i += 256) s += p[i];
  red[threadIdx.x] = s; __syncthreads();
  for (int st = 128; st > 0; st >>= 1) {
    if (threadIdx.x < st) red[threadIdx.x] += red[threadIdx.x + st];
    __syncthreads();
  }
  if (threadIdx.x == 0) ybuf[item] = red[0] * (1.0f / (float)HW);
}

// ------------- ECA 1-D conv(k=3) + sigmoid gate + residual add -------------
__global__ void final_k(const float* __restrict__ x, const float* __restrict__ t4,
                        const float* __restrict__ ybuf, const float* __restrict__ keca,
                        float* __restrict__ out)
{
  long i = blockIdx.x * (long)blockDim.x + threadIdx.x;
  if (i >= (long)BTN * 128 * HW) return;
  long r = i / HW; int c = (int)(r % 128); int bt = (int)(r / 128);
  int n = bt >> 3, t = bt & 7;
  const float* yrow = ybuf + ((long)n * 128 + c) * 8;
  float ym1 = (t > 0) ? yrow[t - 1] : 0.f;
  float y0  = yrow[t];
  float yp1 = (t < 7) ? yrow[t + 1] : 0.f;
  float v   = keca[0] * ym1 + keca[1] * y0 + keca[2] * yp1;
  float sig = 1.0f / (1.0f + expf(-v));
  out[i] = x[i] + t4[i] * sig;
}

// ---------------------------------------------------------------------------
extern "C" void kernel_launch(void* const* d_in, const int* in_sizes, int n_in,
                              void* d_out, int out_size, void* d_ws, size_t ws_size,
                              hipStream_t stream)
{
  (void)in_sizes; (void)n_in; (void)out_size; (void)ws_size;
  const float* x      = (const float*)d_in[0];
  const float* w_redu = (const float*)d_in[1];
  const float* s_redu = (const float*)d_in[2];
  const float* b_redu = (const float*)d_in[3];
  const float* w1a = (const float*)d_in[4],  *s1a = (const float*)d_in[5],  *b1a = (const float*)d_in[6];
  const float* w1b = (const float*)d_in[7],  *s1b = (const float*)d_in[8],  *b1b = (const float*)d_in[9];
  const float* w1c = (const float*)d_in[10], *s1c = (const float*)d_in[11], *b1c = (const float*)d_in[12];
  const float* w1p = (const float*)d_in[13], *s1p = (const float*)d_in[14], *b1p = (const float*)d_in[15];
  const float* w2d = (const float*)d_in[16], *s2d = (const float*)d_in[17], *b2d = (const float*)d_in[18];
  const float* w2p = (const float*)d_in[19], *s2p = (const float*)d_in[20], *b2p = (const float*)d_in[21];
  const float* w3d = (const float*)d_in[22], *s3d = (const float*)d_in[23], *b3d = (const float*)d_in[24];
  const float* w3p = (const float*)d_in[25], *s3p = (const float*)d_in[26], *b3p = (const float*)d_in[27];
  const float* w4d = (const float*)d_in[28], *s4d = (const float*)d_in[29], *b4d = (const float*)d_in[30];
  const float* w4p = (const float*)d_in[31], *s4p = (const float*)d_in[32], *b4p = (const float*)d_in[33];
  const float* w_eca = (const float*)d_in[34];
  float* out = (float*)d_out;

  // workspace layout (floats): total ~26.3M floats (~105 MB)
  float* wsf  = (float*)d_ws;
  float* xn   = wsf;                       // 128*64*784  = 6,422,528
  float* buf6 = xn   + 6422528;            // 128*6*784   =   602,112
  float* bufA = buf6 + 602112;             // 128*128*784 = 12,845,056
  float* bufB = bufA + 12845056;           // 128*64*784  = 6,422,528
  float* ybuf = bufB + 6422528;            // 16*128*8    =    16,384

  const int MT = BTN * HW / 16;            // 6272 pixel tiles
  // 1) reduce conv 128->64 (fp32 WMMA), BN+ReLU
  pconv_f32wmma_k<<<dim3(MT, 4), 32, 0, stream>>>(x, w_redu, s_redu, b_redu, xn, 128, 64);
  // 2) channel L2-norm in place
  l2norm_k<<<(BTN * HW + 255) / 256, 256, 0, stream>>>(xn);
  // 3) correlation + flow_soft, both branches fused
  corrflow_k<<<dim3(HW, 7, NCLIP), 256, 0, stream>>>(xn, buf6);
  // 4) TAM stack
  long e6 = (long)BTN * 6 * HW;
  gconv6_k<<<(int)((e6 + 255) / 256), 256, 0, stream>>>(buf6, w1a, s1a, b1a, bufB);
  gconv6_k<<<(int)((e6 + 255) / 256), 256, 0, stream>>>(bufB, w1b, s1b, b1b, bufA);
  gconv6_k<<<(int)((e6 + 255) / 256), 256, 0, stream>>>(bufA, w1c, s1c, b1c, bufB);
  pconv_f16wmma_k<<<dim3(MT, 1), 32, 0, stream>>>(bufB, w1p, s1p, b1p, bufA, 6, 16);
  long e16 = (long)BTN * 16 * HW;
  dwconv_k<<<(int)((e16 + 255) / 256), 256, 0, stream>>>(bufA, w2d, s2d, b2d, bufB, 16);
  pconv_f16wmma_k<<<dim3(MT, 2), 32, 0, stream>>>(bufB, w2p, s2p, b2p, bufA, 16, 32);
  long e32 = (long)BTN * 32 * HW;
  dwconv_k<<<(int)((e32 + 255) / 256), 256, 0, stream>>>(bufA, w3d, s3d, b3d, bufB, 32);
  pconv_f16wmma_k<<<dim3(MT, 4), 32, 0, stream>>>(bufB, w3p, s3p, b3p, bufA, 32, 64);
  long e64 = (long)BTN * 64 * HW;
  dwconv_k<<<(int)((e64 + 255) / 256), 256, 0, stream>>>(bufA, w4d, s4d, b4d, bufB, 64);
  pconv_f16wmma_k<<<dim3(MT, 8), 32, 0, stream>>>(bufB, w4p, s4p, b4p, bufA, 64, 128);
  // 5) ECA + residual
  eca_mean_k<<<NCLIP * 128 * NSEG, 256, 0, stream>>>(bufA, ybuf);
  long etot = (long)BTN * 128 * HW;
  final_k<<<(int)((etot + 255) / 256), 256, 0, stream>>>(x, bufA, ybuf, w_eca, out);
}
